// KGAdapterLayer_29506425323958
// MI455X (gfx1250) — matrix operand order repaired and checked by reading, not compiled, not run on hardware
//
#include <hip/hip_runtime.h>
#include <hip/hip_bf16.h>

#define N_NODES 50000
#define N_EDGES 600000
#define DDIM    128
#define LN_EPS  1e-5f

typedef __attribute__((ext_vector_type(16))) __bf16 v16bf;
typedef __attribute__((ext_vector_type(8)))  __bf16 v8bf;
typedef __attribute__((ext_vector_type(8)))  float  v8f;

static inline size_t al256(size_t x) { return (x + 255) & ~(size_t)255; }

// ---------------- WMMA fragment helpers (wave32, 16x16x32 bf16) ----------------

__device__ __forceinline__ v16bf a_frag(const __bf16* A, int ld, int mbase, int k0, int lane) {
    // A-matrix 16x32 bf16 layout: lane L row = L&15;
    // lanes 0-15: K = k0+{0..7} and k0+16+{0..7}; lanes 16-31: +8 on both halves.
    const __bf16* p = A + (mbase + (lane & 15)) * ld + k0 + ((lane >> 4) << 3);
    union { v16bf v; v8bf h[2]; } u;
    u.h[0] = *(const v8bf*)(p);
    u.h[1] = *(const v8bf*)(p + 16);
    return u.v;
}

__device__ __forceinline__ v16bf b_frag(const __bf16* Wt, int ldk, int n0, int k0, int lane) {
    // B-matrix 32x16 bf16: lane L column = n0+(L&15); lanes 0-15: K = k0+{0..7} and k0+16+{0..7}; lanes 16-31: +8 on both halves.
    // Wt is the pre-transposed weight: Wt[n][k] contiguous in k.
    const __bf16* p = Wt + (n0 + (lane & 15)) * ldk + k0 + ((lane >> 4) << 3);
    union { v16bf v; v8bf h[2]; } u;
    u.h[0] = *(const v8bf*)(p);
    u.h[1] = *(const v8bf*)(p + 16);
    return u.v;
}

__device__ __forceinline__ v8f wmma_bf16(v16bf a, v16bf b, v8f c) {
    return __builtin_amdgcn_wmma_f32_16x16x32_bf16(false, a, false, b, (short)0, c, false, false);
}

__device__ __forceinline__ float wave_sum(float x) {
    for (int m = 16; m >= 1; m >>= 1) x += __shfl_xor(x, m, 32);
    return x;
}
__device__ __forceinline__ float wave_max(float x) {
    for (int m = 16; m >= 1; m >>= 1) x = fmaxf(x, __shfl_xor(x, m, 32));
    return x;
}

// ---------------- K0: transpose + convert weights to bf16 ----------------
__global__ void cvt_weights_kernel(const float* __restrict__ Wq, const float* __restrict__ Wk,
                                   const float* __restrict__ Wv, const float* __restrict__ Wo,
                                   const float* __restrict__ W1, const float* __restrict__ W2,
                                   __bf16* __restrict__ Wqkvt, __bf16* __restrict__ W1t,
                                   __bf16* __restrict__ W2t,   __bf16* __restrict__ Wot) {
    int i = blockIdx.x * blockDim.x + threadIdx.x;
    if (i < 384 * 128) {
        int j = i >> 7, k = i & 127;
        const float* W = (j < 128) ? Wq : (j < 256) ? Wk : Wv;
        Wqkvt[i] = (__bf16)W[k * 128 + (j & 127)];
    }
    if (i < 128 * 384) {
        int n = i / 384, k = i - n * 384;
        W1t[i] = (__bf16)W1[k * 128 + n];
    }
    if (i < 128 * 128) {
        int n = i >> 7, k = i & 127;
        W2t[i] = (__bf16)W2[k * 128 + n];
        Wot[i] = (__bf16)Wo[k * 128 + n];
    }
}

// ---------------- K1: fused QKV GEMM + emit bf16 node matrix ----------------
// grid = (ceil(N/64), 3); blockIdx.y picks 8 of the 24 column tiles, wave w -> one tile.
__global__ void qkv_gemm_kernel(const float* __restrict__ nodes, const __bf16* __restrict__ Wqkvt,
                                float* __restrict__ q, float* __restrict__ kout,
                                float* __restrict__ vout, __bf16* __restrict__ node_bf16) {
    __shared__ __bf16 A[64][136];
    const int r0 = blockIdx.x * 64;
    const int t  = threadIdx.x;
    const bool emit_bf16 = (blockIdx.y == 0);

    for (int i = 0; i < 8; ++i) {
        int f = t + i * 256;              // 0..2047 float4 slots, [64][32]
        int row = f >> 5, c4 = f & 31;
        int grow = r0 + row;
        float4 val = make_float4(0.f, 0.f, 0.f, 0.f);
        if (grow < N_NODES) val = ((const float4*)nodes)[grow * 32 + c4];
        __bf16* d = &A[row][c4 * 4];
        d[0] = (__bf16)val.x; d[1] = (__bf16)val.y; d[2] = (__bf16)val.z; d[3] = (__bf16)val.w;
        if (emit_bf16 && grow < N_NODES) {
            __bf16* g = node_bf16 + grow * 128 + c4 * 4;
            g[0] = (__bf16)val.x; g[1] = (__bf16)val.y; g[2] = (__bf16)val.z; g[3] = (__bf16)val.w;
        }
    }
    __syncthreads();

    const int w = t >> 5, lane = t & 31;
    const int ct = w + blockIdx.y * 8;    // 0..23, fixed per wave
    v8f acc[4] = {};
    #pragma unroll 1
    for (int kk = 0; kk < 128; kk += 32) {
        v16bf b = b_frag(Wqkvt, 128, ct * 16, kk, lane);        // 1 global B frag...
        for (int rt = 0; rt < 4; ++rt) {                        // ...amortized over 4 WMMAs
            v16bf a = a_frag(&A[0][0], 136, rt * 16, kk, lane);
            acc[rt] = wmma_bf16(a, b, acc[rt]);
        }
    }
    int n0 = ct * 16;
    float* outp; int col;
    if (n0 < 128)       { outp = q;    col = n0; }
    else if (n0 < 256)  { outp = kout; col = n0 - 128; }
    else                { outp = vout; col = n0 - 256; }
    col += (lane & 15);
    for (int rt = 0; rt < 4; ++rt) {
        for (int r = 0; r < 8; ++r) {
            int row = r0 + rt * 16 + r + ((lane >> 4) << 3);
            if (row < N_NODES) outp[row * 128 + col] = acc[rt][r];
        }
    }
}

// ---------------- K2: per-edge attention scores (wave per edge) ----------------
__global__ void edge_scores_kernel(const float* __restrict__ q, const float* __restrict__ kx,
                                   const float* __restrict__ edges, const int* __restrict__ src,
                                   const int* __restrict__ dst, float* __restrict__ scores) {
    int w = threadIdx.x >> 5, lane = threadIdx.x & 31;
    int e = blockIdx.x * 8 + w;
    if (e >= N_EDGES) return;
    int s = src[e], d = dst[e];
    float4 qv = ((const float4*)q)[d * 32 + lane];
    float4 kv = ((const float4*)kx)[s * 32 + lane];
    float4 ev = ((const float4*)edges)[e * 32 + lane];
    float p = qv.x * (kv.x + ev.x) + qv.y * (kv.y + ev.y) +
              qv.z * (kv.z + ev.z) + qv.w * (kv.w + ev.w);
    p = wave_sum(p);
    if (lane == 0) scores[e] = p * 0.08838834764831845f;   // 1/sqrt(128)
}

// ---------------- K3: CSR begin[] from sorted dst_idx ----------------
__global__ void seg_begin_kernel(const int* __restrict__ dst, int* __restrict__ begin) {
    int e = blockIdx.x * blockDim.x + threadIdx.x;
    if (e >= N_EDGES) return;
    int de = dst[e];
    if (e == 0) {
        for (int n = 0; n <= de; ++n) begin[n] = 0;
    } else {
        int dp = dst[e - 1];
        for (int n = dp + 1; n <= de; ++n) begin[n] = e;
    }
    if (e == N_EDGES - 1) {
        for (int n = de + 1; n <= N_NODES; ++n) begin[n] = N_EDGES;
    }
}

// ---------------- K4: segment softmax + aggregation (wave per node) ----------------
__global__ void node_agg_kernel(const float* __restrict__ scores, const int* __restrict__ begin,
                                const int* __restrict__ src, const float* __restrict__ vx,
                                const float* __restrict__ edges, float* __restrict__ agg) {
    int w = threadIdx.x >> 5, lane = threadIdx.x & 31;
    int n = blockIdx.x * 8 + w;
    if (n >= N_NODES) return;
    int b0 = begin[n], b1 = begin[n + 1];
    float4 acc = make_float4(0.f, 0.f, 0.f, 0.f);
    if (b0 < b1) {
        float m = -3.4e38f;
        for (int e = b0 + lane; e < b1; e += 32) m = fmaxf(m, scores[e]);
        m = wave_max(m);
        float denom = 0.f;
        for (int e = b0; e < b1; ++e) {
            float ex = __expf(scores[e] - m);      // uniform across lanes
            denom += ex;
            int s = src[e];
            float4 vv = ((const float4*)vx)[s * 32 + lane];
            float4 ee = ((const float4*)edges)[e * 32 + lane];
            acc.x += ex * (vv.x + ee.x); acc.y += ex * (vv.y + ee.y);
            acc.z += ex * (vv.z + ee.z); acc.w += ex * (vv.w + ee.w);
        }
        float inv = 1.f / denom;
        acc.x *= inv; acc.y *= inv; acc.z *= inv; acc.w *= inv;
    }
    ((float4*)agg)[n * 32 + lane] = acc;
}

// ---------------- K5: h = node + agg @ Wo (WMMA) ----------------
__global__ void wo_gemm_kernel(const float* __restrict__ agg, const float* __restrict__ nodes,
                               const __bf16* __restrict__ Wot, float* __restrict__ h) {
    __shared__ __bf16 A[64][136];
    const int r0 = blockIdx.x * 64;
    const int t  = threadIdx.x;
    for (int i = 0; i < 8; ++i) {
        int f = t + i * 256;
        int row = f >> 5, c4 = f & 31;
        int grow = r0 + row;
        float4 val = make_float4(0.f, 0.f, 0.f, 0.f);
        if (grow < N_NODES) val = ((const float4*)agg)[grow * 32 + c4];
        __bf16* d = &A[row][c4 * 4];
        d[0] = (__bf16)val.x; d[1] = (__bf16)val.y; d[2] = (__bf16)val.z; d[3] = (__bf16)val.w;
    }
    __syncthreads();

    const int w = t >> 5, lane = t & 31;
    const int ct = w;                      // 8 column tiles, one per wave
    v8f acc[4] = {};
    #pragma unroll 1
    for (int kk = 0; kk < 128; kk += 32) {
        v16bf b = b_frag(Wot, 128, ct * 16, kk, lane);
        for (int rt = 0; rt < 4; ++rt) {
            v16bf a = a_frag(&A[0][0], 136, rt * 16, kk, lane);
            acc[rt] = wmma_bf16(a, b, acc[rt]);
        }
    }
    int col = ct * 16 + (lane & 15);
    for (int rt = 0; rt < 4; ++rt) {
        for (int r = 0; r < 8; ++r) {
            int row = r0 + rt * 16 + r + ((lane >> 4) << 3);
            if (row < N_NODES) h[row * 128 + col] = nodes[row * 128 + col] + acc[rt][r];
        }
    }
}

// ---------------- K6: LayerNorm (wave per row) ----------------
__global__ void layer_norm_kernel(const float* __restrict__ h, const float* __restrict__ gamma,
                                  const float* __restrict__ beta, float* __restrict__ out) {
    int w = threadIdx.x >> 5, lane = threadIdx.x & 31;
    int n = blockIdx.x * 8 + w;
    if (n >= N_NODES) return;
    float4 x = ((const float4*)h)[n * 32 + lane];
    float mu = wave_sum(x.x + x.y + x.z + x.w) * (1.f / 128.f);
    float4 dx = make_float4(x.x - mu, x.y - mu, x.z - mu, x.w - mu);
    float var = wave_sum(dx.x * dx.x + dx.y * dx.y + dx.z * dx.z + dx.w * dx.w) * (1.f / 128.f);
    float rinv = __frsqrt_rn(var + LN_EPS);
    float4 g = ((const float4*)gamma)[lane];
    float4 b = ((const float4*)beta)[lane];
    float4 res;
    res.x = dx.x * rinv * g.x + b.x;
    res.y = dx.y * rinv * g.y + b.y;
    res.z = dx.z * rinv * g.z + b.z;
    res.w = dx.w * rinv * g.w + b.w;
    ((float4*)out)[n * 32 + lane] = res;
}

// ---------------- K7: triplet MLP (64 edges per block, WMMA x2) ----------------
__global__ void mlp_kernel(const __bf16* __restrict__ node_bf16, const float* __restrict__ edges,
                           const int* __restrict__ src, const int* __restrict__ dst,
                           const __bf16* __restrict__ W1t, const float* __restrict__ b1,
                           const __bf16* __restrict__ W2t, const float* __restrict__ b2,
                           float* __restrict__ out_fused) {
    __shared__ __bf16 A[64][384];      // triplet tile [dst | edge | src], bf16
    __shared__ int sidx[64], didx[64];
    const int e0 = blockIdx.x * 64;
    const int t  = threadIdx.x;

    if (t < 64) {
        int e = e0 + t;
        sidx[t] = (e < N_EDGES) ? src[e] : 0;
        didx[t] = (e < N_EDGES) ? dst[e] : 0;
    }
    __syncthreads();

    {   // stage: thread t handles row t/4, 32-element chunk t%4
        int row = t >> 2, c = t & 3;
        int e = e0 + row;
        bool valid = (e < N_EDGES);
        uint4 z = make_uint4(0u, 0u, 0u, 0u);
        const uint4* pd = (const uint4*)(node_bf16 + (size_t)didx[row] * 128 + c * 32);
        const uint4* ps = (const uint4*)(node_bf16 + (size_t)sidx[row] * 128 + c * 32);
        uint4 d0 = valid ? pd[0] : z, d1 = valid ? pd[1] : z;
        uint4 s0 = valid ? ps[0] : z, s1 = valid ? ps[1] : z;
        *(uint4*)&A[row][c * 32]           = d0; *(uint4*)&A[row][c * 32 + 8]       = d1;
        *(uint4*)&A[row][256 + c * 32]     = s0; *(uint4*)&A[row][256 + c * 32 + 8] = s1;
        const float4* pe = (const float4*)(edges + (size_t)e * 128 + c * 32);
        for (int j = 0; j < 8; ++j) {
            float4 val = valid ? pe[j] : make_float4(0.f, 0.f, 0.f, 0.f);
            __bf16* d = &A[row][128 + c * 32 + j * 4];
            d[0] = (__bf16)val.x; d[1] = (__bf16)val.y; d[2] = (__bf16)val.z; d[3] = (__bf16)val.w;
        }
    }
    __syncthreads();

    const int w = t >> 5, lane = t & 31;
    const int ct = w;                          // 8 column tiles, one per wave

    // GEMM1: hidden = relu(trip @ W1 + b1); 12 K-steps, 1 B load : 4 WMMA
    v8f acc[4] = {};
    #pragma unroll 1
    for (int kk = 0; kk < 384; kk += 32) {
        v16bf b = b_frag(W1t, 384, ct * 16, kk, lane);
        for (int rt = 0; rt < 4; ++rt) {
            v16bf a = a_frag(&A[0][0], 384, rt * 16, kk, lane);
            acc[rt] = wmma_bf16(a, b, acc[rt]);
        }
    }
    __syncthreads();                           // all waves done reading A

    __bf16 (*H)[128] = (__bf16 (*)[128])&A[0][0];   // reuse LDS for hidden
    {
        float bias = b1[ct * 16 + (lane & 15)];
        for (int rt = 0; rt < 4; ++rt) {
            for (int r = 0; r < 8; ++r) {
                float hv = acc[rt][r] + bias;
                hv = hv > 0.f ? hv : 0.f;
                H[rt * 16 + r + ((lane >> 4) << 3)][ct * 16 + (lane & 15)] = (__bf16)hv;
            }
        }
    }
    __syncthreads();

    // GEMM2: fused = hidden @ W2 + b2
    v8f acc2[4] = {};
    #pragma unroll 1
    for (int kk = 0; kk < 128; kk += 32) {
        v16bf b = b_frag(W2t, 128, ct * 16, kk, lane);
        for (int rt = 0; rt < 4; ++rt) {
            v16bf a = a_frag(&H[0][0], 128, rt * 16, kk, lane);
            acc2[rt] = wmma_bf16(a, b, acc2[rt]);
        }
    }
    {
        float bias = b2[ct * 16 + (lane & 15)];
        int col = ct * 16 + (lane & 15);
        for (int rt = 0; rt < 4; ++rt) {
            for (int r = 0; r < 8; ++r) {
                int e = e0 + rt * 16 + r + ((lane >> 4) << 3);
                if (e < N_EDGES) out_fused[(size_t)e * 128 + col] = acc2[rt][r] + bias;
            }
        }
    }
}

// ---------------- host launch ----------------
extern "C" void kernel_launch(void* const* d_in, const int* in_sizes, int n_in,
                              void* d_out, int out_size, void* d_ws, size_t ws_size,
                              hipStream_t stream) {
    const float* node_reps = (const float*)d_in[0];
    const float* edge_reps = (const float*)d_in[1];
    const int*   src_idx   = (const int*)d_in[2];
    const int*   dst_idx   = (const int*)d_in[3];
    const float* Wq        = (const float*)d_in[4];
    const float* Wk        = (const float*)d_in[5];
    const float* Wv        = (const float*)d_in[6];
    const float* Wo        = (const float*)d_in[7];
    const float* ln_gamma  = (const float*)d_in[8];
    const float* ln_beta   = (const float*)d_in[9];
    const float* W1        = (const float*)d_in[10];
    const float* b1        = (const float*)d_in[11];
    const float* W2        = (const float*)d_in[12];
    const float* b2        = (const float*)d_in[13];

    float* out_updated = (float*)d_out;
    float* out_fused   = out_updated + (size_t)N_NODES * 128;

    // workspace carve (256B aligned)
    char* base = (char*)d_ws;
    size_t off = 0;
    float* q     = (float*)(base + off);  off += al256((size_t)N_NODES * 128 * 4);
    float* kbuf  = (float*)(base + off);  off += al256((size_t)N_NODES * 128 * 4);
    float* vbuf  = (float*)(base + off);  off += al256((size_t)N_NODES * 128 * 4);
    float* agg   = (float*)(base + off);  off += al256((size_t)N_NODES * 128 * 4);
    float* hbuf  = (float*)(base + off);  off += al256((size_t)N_NODES * 128 * 4);
    float* scores= (float*)(base + off);  off += al256((size_t)N_EDGES * 4);
    int*   begin = (int*)(base + off);    off += al256((size_t)(N_NODES + 1) * 4);
    __bf16* node_bf16 = (__bf16*)(base + off); off += al256((size_t)N_NODES * 128 * 2);
    __bf16* Wqkvt = (__bf16*)(base + off); off += al256((size_t)384 * 128 * 2);
    __bf16* W1t   = (__bf16*)(base + off); off += al256((size_t)128 * 384 * 2);
    __bf16* W2t   = (__bf16*)(base + off); off += al256((size_t)128 * 128 * 2);
    __bf16* Wot   = (__bf16*)(base + off); off += al256((size_t)128 * 128 * 2);
    (void)ws_size; (void)in_sizes; (void)n_in; (void)out_size;

    cvt_weights_kernel<<<192, 256, 0, stream>>>(Wq, Wk, Wv, Wo, W1, W2, Wqkvt, W1t, W2t, Wot);
    qkv_gemm_kernel<<<dim3((N_NODES + 63) / 64, 3), 256, 0, stream>>>(node_reps, Wqkvt, q, kbuf, vbuf, node_bf16);
    edge_scores_kernel<<<(N_EDGES + 7) / 8, 256, 0, stream>>>(q, kbuf, edge_reps, src_idx, dst_idx, scores);
    seg_begin_kernel<<<(N_EDGES + 255) / 256, 256, 0, stream>>>(dst_idx, begin);
    node_agg_kernel<<<(N_NODES + 7) / 8, 256, 0, stream>>>(scores, begin, src_idx, vbuf, edge_reps, agg);
    wo_gemm_kernel<<<(N_NODES + 63) / 64, 256, 0, stream>>>(agg, node_reps, Wot, hbuf);
    layer_norm_kernel<<<(N_NODES + 7) / 8, 256, 0, stream>>>(hbuf, ln_gamma, ln_beta, out_updated);
    mlp_kernel<<<(N_EDGES + 63) / 64, 256, 0, stream>>>(node_bf16, edge_reps, src_idx, dst_idx,
                                                        W1t, b1, W2t, b2, out_fused);
}
